// Model_4294967296110
// MI455X (gfx1250) — matrix operand order repaired and checked by reading, not compiled, not run on hardware
//
#include <hip/hip_runtime.h>
#include <hip/hip_bf16.h>
#include <cstddef>
#include <cstdint>

typedef __attribute__((ext_vector_type(16))) __bf16 v16bf;
typedef __attribute__((ext_vector_type(8)))  float  v8f;
typedef __attribute__((ext_vector_type(4)))  unsigned u32x4;
typedef __attribute__((ext_vector_type(8)))  int      i32x8;
typedef __attribute__((ext_vector_type(4)))  int      i32x4;

// Address-space qualified vector pointee types for the async LDS builtins
typedef __attribute__((address_space(1))) i32x4 as1_v4i;   // global
typedef __attribute__((address_space(3))) i32x4 as3_v4i;   // LDS

#define WMMA_BF16(a, b, c) \
  __builtin_amdgcn_wmma_f32_16x16x32_bf16(false, (a), false, (b), (short)0, (c), false, false)

// Feature probes (keep compile robust across toolchains)
#if defined(__has_builtin)
#  if __has_builtin(__builtin_amdgcn_tensor_load_to_lds)
#    define HAVE_TDM 1
#  endif
#  if __has_builtin(__builtin_amdgcn_global_load_async_to_lds_b128)
#    define HAVE_ASYNC_LD 1
#  endif
#  if __has_builtin(__builtin_amdgcn_global_store_async_from_lds_b128)
#    define HAVE_ASYNC_ST 1
#  endif
#  if __has_builtin(__builtin_amdgcn_s_wait_asynccnt)
#    define HAVE_WAIT_ASYNC 1
#  endif
#  if __has_builtin(__builtin_amdgcn_tanhf)
#    define HAVE_TANH 1
#  endif
#endif

// Problem constants
#define Bq   16      // batch == WMMA M
#define Hd   512     // hidden
#define G3   1536    // 3*H gate width
#define Tlen 201
#define NC   80
#define NBLK 16      // workgroups (one cluster)
#define NTHR 192     // 6 waves, 1 gate tile each

// Workspace layout (bytes)
#define OFF_BAR   0u
#define OFF_WHH   256u                           // 96*16*32*16 bf16 = 1572864
#define OFF_WIH   (OFF_WHH + 1572864u)
#define OFF_WPJ   (OFF_WIH + 1572864u)           // 5*16*32*16 bf16 = 81920
#define OFF_GI    (OFF_WPJ + 81920u)             // 201*16*512*3 f32 = 19759104
#define OFF_ABUF  (OFF_GI + 19759104u)           // 201*16*32*16 bf16 = 3293184
#define OFF_HPK   (OFF_ABUF + 3293184u)          // 2 x 16KB A-packed hidden (double buffer)

static __device__ __forceinline__ float sigm(float x) {
  return __builtin_amdgcn_rcpf(1.0f + __expf(-x));
}
static __device__ __forceinline__ float tanh_fast(float x) {
#if HAVE_TANH
  return __builtin_amdgcn_tanhf(x);
#else
  return tanhf(x);
#endif
}

static __device__ __forceinline__ void wait_async0() {
#if HAVE_WAIT_ASYNC
  __builtin_amdgcn_s_wait_asynccnt(0);
#elif HAVE_ASYNC_LD || HAVE_ASYNC_ST
  asm volatile("s_wait_asynccnt 0" ::: "memory");
#endif
}

// ---------------------------------------------------------------------------
// Pack W[N][512] (f32) into per-lane WMMA B-fragments (bf16):
//   pack[((tile*16+ks)*32 + lane)*16 + e] = W[tile*16 + (lane&15)][ks*32 + 8*(lane>>4) + (e&7) + 16*(e>>3)]
// ---------------------------------------------------------------------------
__global__ void pack_weights_kernel(const float* __restrict__ whh,
                                    const float* __restrict__ wih,
                                    const float* __restrict__ wpj,
                                    __bf16* __restrict__ pwhh,
                                    __bf16* __restrict__ pwih,
                                    __bf16* __restrict__ ppj) {
  size_t idx = (size_t)blockIdx.x * blockDim.x + threadIdx.x;
  const size_t N1 = 96ull * 16 * 32 * 16;
  const size_t N2 = N1;
  const size_t N3 = 5ull * 16 * 32 * 16;
  const float* src; __bf16* dst; size_t li;
  if (idx < N1)            { src = whh; dst = pwhh; li = idx; }
  else if (idx < N1 + N2)  { src = wih; dst = pwih; li = idx - N1; }
  else if (idx < N1+N2+N3) { src = wpj; dst = ppj;  li = idx - N1 - N2; }
  else return;
  int e    = (int)(li & 15);
  int lane = (int)((li >> 4) & 31);
  int ks   = (int)((li >> 9) & 15);
  int tile = (int)(li >> 13);
  dst[li] = (__bf16)src[(size_t)(tile * 16 + (lane & 15)) * Hd + ks * 32 + 8 * (lane >> 4) + (e & 7) + 16 * (e >> 3)];
}

// ---------------------------------------------------------------------------
// Persistent GRU kernel: 16 workgroups (one cluster), 6 waves each.
// w_hh B-fragments register-resident; per cell: 16 WMMAs + combine + one
// cluster/grid barrier + one TDM refill of the A-fragment LDS tile.
// ---------------------------------------------------------------------------
__global__ void __launch_bounds__(NTHR, 1)
gru_kernel(const float* __restrict__ feat,
           const float* __restrict__ b_ih,
           const float* __restrict__ b_hh,
           const float* __restrict__ embed,
           const __bf16* __restrict__ whhp,
           const __bf16* __restrict__ wihp,
           float* __restrict__ GIp,
           __bf16* __restrict__ abufp,
           __bf16* __restrict__ hpk,       // 2 x 8192 bf16: A-packed hidden (double buffer)
           unsigned* __restrict__ barp) {
  const int r    = blockIdx.x;          // 0..15 : owns hidden cols [32r, 32r+32)
  const int wv   = threadIdx.x >> 5;    // 0..5
  const int lane = threadIdx.x & 31;
  // wave -> gate tile: {2r,2r+1 (reset), 32+2r,33+2r (update), 64+2r,65+2r (new)}
  const int tile = (wv >> 1) * 32 + 2 * r + (wv & 1);
  const int gate = tile >> 5;           // 0=r, 1=z, 2=n

  __shared__ alignas(64) __bf16 Apack[16][32][16];  // A-fragments of current h (16 KB)
  __shared__ float ghLDS[6][16][16];
  __shared__ float Hown[Bq][32];

  unsigned* bcnt = barp;
  unsigned* bgen = barp + 1;

  // ---- w_hh B-fragments: VGPR-resident for the whole kernel (16 KB/wave)
  v16bf breg[16];
#pragma unroll
  for (int ks = 0; ks < 16; ++ks)
    breg[ks] = *(const v16bf*)(whhp + (((size_t)tile * 16 + ks) * 32 + lane) * 16);

  // n-gate hidden bias (inside the r* product) folded into ghLDS at store time
  const float bh_n = (wv >= 4) ? b_hh[tile * 16 + (lane & 15)] : 0.0f;

#if HAVE_TDM
  // TDM descriptors: contiguous 16 KB (16 rows x 256 x 4B) global -> LDS Apack
  const unsigned lds_addr = (unsigned)(uintptr_t)(void*)&Apack[0][0][0];
  const unsigned long long ga0 = (unsigned long long)(uintptr_t)hpk;
  u32x4 g0a{}, g0b{};
  {
    g0a[0] = 1u;                                   // count=1, user descriptor
    g0a[1] = lds_addr;
    g0a[2] = (unsigned)(ga0 & 0xffffffffu);
    g0a[3] = (unsigned)((ga0 >> 32) & 0x1ffffffu) | (2u << 30);  // type=2
    unsigned long long ga1 = ga0 + 16384ull;
    g0b[0] = 1u;
    g0b[1] = lds_addr;
    g0b[2] = (unsigned)(ga1 & 0xffffffffu);
    g0b[3] = (unsigned)((ga1 >> 32) & 0x1ffffffu) | (2u << 30);
  }
  i32x8 g1d{};
  g1d[0] = (int)(2u << 16);      // data_size = 4 bytes
  g1d[1] = (int)(256u << 16);    // tensor_dim0 = 256
  g1d[2] = (int)(16u << 16);     // tensor_dim1 = 16
  g1d[3] = (int)(256u << 16);    // tile_dim0 = 256
  g1d[4] = 16;                   // tile_dim1 = 16
  g1d[5] = 256;                  // tensor_dim0_stride = 256
  i32x4 gz4{};
#endif

  // Refill Apack LDS from the A-packed hidden staging buffer (parity-selected).
  auto refillA = [&](unsigned par) {
#if HAVE_TDM
    if (wv == 0) {
      u32x4 g0 = par ? g0b : g0a;
#if defined(__clang_major__) && __clang_major__ >= 23
      i32x8 g4{};
      __builtin_amdgcn_tensor_load_to_lds(g0, g1d, gz4, gz4, g4, 0);
#else
      __builtin_amdgcn_tensor_load_to_lds(g0, g1d, gz4, gz4, 0);
#endif
      __builtin_amdgcn_s_wait_tensorcnt((short)0);
    }
    __syncthreads();
#elif HAVE_ASYNC_LD
    {
      const char* src = (const char*)hpk + (size_t)par * 16384u;
      char* dst = (char*)&Apack[0][0][0];
      for (int off = threadIdx.x * 16; off < 16384; off += NTHR * 16)
        __builtin_amdgcn_global_load_async_to_lds_b128(
            (as1_v4i*)(uintptr_t)(src + off),
            (as3_v4i*)(uintptr_t)(dst + off), 0, 0);
      wait_async0();
    }
    __syncthreads();
#else
    {
      const int4* src = (const int4*)((const char*)hpk + (size_t)par * 16384u);
      int4* dst = (int4*)&Apack[0][0][0];
      for (int v = threadIdx.x; v < 1024; v += NTHR) dst[v] = src[v];
    }
    __syncthreads();
#endif
  };

  // Initial pack of a [16][512] f32 matrix (or broadcast row) into Apack.
  auto packA = [&](const float* __restrict__ hsrc, bool bcast) {
    for (int idx = threadIdx.x; idx < Bq * Hd; idx += NTHR) {
      int m = idx >> 9, K = idx & 511;
      float v = bcast ? hsrc[K] : hsrc[m * Hd + K];
      int ks = K >> 5, kk = K & 31;
      int b4 = kk >> 4, rem = kk & 15;
      int khalf = rem >> 3, r8 = rem & 7;
      int p = (r8 >> 1) + (b4 ? 4 : 0);
      Apack[ks][khalf * 16 + m][2 * p + (r8 & 1)] = (__bf16)v;
    }
  };

  // GI[tt] = Apack @ w_ih^T + b_ih (+ b_hh for r/z gates, which sit outside
  // the r* product). Layout gate-minor: GI[t][m][col][3] so the combine reads
  // one contiguous b96 per hidden column.
  auto computeGI = [&](int tt) {
    __builtin_prefetch(wihp + ((size_t)tile * 16) * 32 * 16, 0, 1);
    v8f acc = {};
#pragma unroll
    for (int ks = 0; ks < 16; ++ks) {
      v16bf a = *(const v16bf*)&Apack[ks][lane][0];
      v16bf b = *(const v16bf*)(wihp + (((size_t)tile * 16 + ks) * 32 + lane) * 16);
      acc = WMMA_BF16(a, b, acc);
    }
    int n    = tile * 16 + (lane & 15);            // gate column in [0,1536)
    int colh = (tile & 31) * 16 + (lane & 15);     // hidden column in [0,512)
    int mb   = (lane >> 4) * 8;
    float bias = b_ih[n] + (gate < 2 ? b_hh[n] : 0.0f);
#pragma unroll
    for (int rr = 0; rr < 8; ++rr)
      GIp[(((size_t)tt * Bq + mb + rr) * Hd + colh) * 3 + gate] = acc[rr] + bias;
  };

  // Persist current Apack (== buf[tt]) into the A-history; block writes kstep slice r.
  auto writeABUF = [&](int tt) {
    __bf16* dst = abufp + (((size_t)tt * 16 + r) * 32) * 16;   // 512 bf16 = 1 KB
#if HAVE_ASYNC_ST
    if (threadIdx.x < 64)
      __builtin_amdgcn_global_store_async_from_lds_b128(
          (as1_v4i*)(uintptr_t)((char*)dst + threadIdx.x * 16),
          (as3_v4i*)(uintptr_t)((char*)&Apack[r][0][0] + threadIdx.x * 16), 0, 0);
    wait_async0();   // LDS source must be stable before next refill overwrites it
#else
    for (int idx = threadIdx.x; idx < 512; idx += NTHR) {
      int l2 = idx >> 4, e = idx & 15;
      dst[idx] = Apack[r][l2][e];
    }
#endif
  };

  // Cluster HW barrier (fast when cluster-dispatched, NOP otherwise) backed by
  // a sense-reversing L2 atomic barrier that is always correct.
  auto grid_barrier = [&]() {
    __syncthreads();
    __builtin_amdgcn_s_cluster_barrier();
    if (threadIdx.x == 0) {
      unsigned g = __hip_atomic_load(bgen, __ATOMIC_RELAXED, __HIP_MEMORY_SCOPE_AGENT);
      unsigned a = __hip_atomic_fetch_add(bcnt, 1u, __ATOMIC_ACQ_REL, __HIP_MEMORY_SCOPE_AGENT);
      if (a == NBLK - 1u) {
        __hip_atomic_store(bcnt, 0u, __ATOMIC_RELAXED, __HIP_MEMORY_SCOPE_AGENT);
        __hip_atomic_fetch_add(bgen, 1u, __ATOMIC_RELEASE, __HIP_MEMORY_SCOPE_AGENT);
      } else {
        while (__hip_atomic_load(bgen, __ATOMIC_ACQUIRE, __HIP_MEMORY_SCOPE_AGENT) == g)
          __builtin_amdgcn_s_sleep(1);
      }
    }
    __syncthreads();
  };

  // ---- init: buf[0] = embed[0] broadcast; GI[0]; then h = feat
  packA(embed, true);
  __syncthreads();
  computeGI(0);
  writeABUF(0);
  __syncthreads();
  packA(feat, false);
  for (int idx = threadIdx.x; idx < 512; idx += NTHR) {
    int m = idx >> 5, jj = idx & 31;
    Hown[m][jj] = feat[m * Hd + r * 32 + jj];
  }
  __syncthreads();

  unsigned cell = 0;
  for (int t = 0; t < Tlen - 1; ++t) {
    for (int i = 0; i <= t; ++i, ++cell) {
      // --- gh = h @ w_hh^T for my tile (weights in registers, A from LDS)
      v8f acc = {};
#pragma unroll
      for (int ks = 0; ks < 16; ++ks) {
        v16bf a = *(const v16bf*)&Apack[ks][lane][0];
        acc = WMMA_BF16(a, breg[ks], acc);
      }
      {
        int n = lane & 15, mb = (lane >> 4) * 8;
#pragma unroll
        for (int rr = 0; rr < 8; ++rr) ghLDS[wv][mb + rr][n] = acc[rr] + bh_n;
      }
      __syncthreads();

      // --- nonlinear combine for own 32 hidden cols; publish new h directly
      //     in bf16 A-fragment layout (double-buffered global staging).
      __bf16* hbw = hpk + (size_t)(cell & 1u) * 8192u;
      for (int idx = threadIdx.x; idx < 512; idx += NTHR) {
        int m = idx >> 5, jj = idx & 31;
        int jl = jj >> 4, n = jj & 15;
        int col = r * 32 + jj;
        const float* gi = GIp + (((size_t)i * Bq + m) * Hd + col) * 3;  // b96
        float rg = sigm(gi[0] + ghLDS[0 + jl][m][n]);
        float zg = sigm(gi[1] + ghLDS[2 + jl][m][n]);
        float ng = tanh_fast(gi[2] + rg * ghLDS[4 + jl][m][n]);  // bias pre-folded
        float hp = Hown[m][jj];
        float hn = (1.0f - zg) * ng + zg * hp;
        Hown[m][jj] = hn;
        // scatter into A-fragment layout: (ks, lane2, e) from (m, col)
        int ks = col >> 5, kk = col & 31;
        int b4 = kk >> 4, rem = kk & 15;
        int khalf = rem >> 3, r8 = rem & 7;
        int p = (r8 >> 1) + (b4 ? 4 : 0);
        hbw[((ks * 32) + (khalf * 16 + m)) * 16 + 2 * p + (r8 & 1)] = (__bf16)hn;
      }

      grid_barrier();            // all 16 col-slices of packed new h visible

      refillA(cell & 1u);        // one TDM tensor_load_to_lds (16 KB) + wait
    }
    // end of outer step t: Apack holds h_final == buf[t+1]
    writeABUF(t + 1);
    computeGI(t + 1);            // block-local; no extra barrier needed
  }
}

// ---------------------------------------------------------------------------
// logits[b][c][t] = buf[t] @ proj_w^T + proj_b  — embarrassingly parallel WMMA
// ---------------------------------------------------------------------------
__global__ void logits_kernel(const __bf16* __restrict__ abufp,
                              const __bf16* __restrict__ ppj,
                              const float* __restrict__ proj_b,
                              float* __restrict__ out) {
  int t  = blockIdx.x / 5;
  int ct = blockIdx.x % 5;
  int lane = threadIdx.x;
  v8f acc = {};
#pragma unroll
  for (int ks = 0; ks < 16; ++ks) {
    v16bf a = *(const v16bf*)(abufp + (((size_t)t * 16 + ks) * 32 + lane) * 16);
    v16bf b = *(const v16bf*)(ppj   + (((size_t)ct * 16 + ks) * 32 + lane) * 16);
    acc = WMMA_BF16(a, b, acc);
  }
  int n  = ct * 16 + (lane & 15);
  int mb = (lane >> 4) * 8;
  float pb = proj_b[n];
#pragma unroll
  for (int rr = 0; rr < 8; ++rr)
    out[(((size_t)(mb + rr)) * NC + n) * Tlen + t] = acc[rr] + pb;
}

extern "C" void kernel_launch(void* const* d_in, const int* in_sizes, int n_in,
                              void* d_out, int out_size, void* d_ws, size_t ws_size,
                              hipStream_t stream) {
  (void)in_sizes; (void)n_in; (void)out_size; (void)ws_size;
  const float* feat   = (const float*)d_in[0];
  const float* w_ih   = (const float*)d_in[1];
  const float* w_hh   = (const float*)d_in[2];
  const float* b_ih   = (const float*)d_in[3];
  const float* b_hh   = (const float*)d_in[4];
  const float* embed  = (const float*)d_in[5];
  const float* proj_w = (const float*)d_in[6];
  const float* proj_b = (const float*)d_in[7];
  float* out = (float*)d_out;

  char* ws = (char*)d_ws;
  unsigned* bar  = (unsigned*)(ws + OFF_BAR);
  __bf16*   pwhh = (__bf16*)(ws + OFF_WHH);
  __bf16*   pwih = (__bf16*)(ws + OFF_WIH);
  __bf16*   ppj  = (__bf16*)(ws + OFF_WPJ);
  float*    GIp  = (float*)(ws + OFF_GI);
  __bf16*   ab   = (__bf16*)(ws + OFF_ABUF);
  __bf16*   hpkp = (__bf16*)(ws + OFF_HPK);

  (void)hipMemsetAsync(ws + OFF_BAR, 0, 256, stream);

  {
    const size_t total = 96ull*16*32*16 * 2 + 5ull*16*32*16;
    int blocks = (int)((total + 255) / 256);
    pack_weights_kernel<<<blocks, 256, 0, stream>>>(w_hh, w_ih, proj_w, pwhh, pwih, ppj);
  }

  gru_kernel<<<NBLK, NTHR, 0, stream>>>(feat, b_ih, b_hh, embed,
                                        pwhh, pwih, GIp, ab, hpkp, bar);

  logits_kernel<<<Tlen * 5, 32, 0, stream>>>(ab, ppj, proj_b, out);
}
